// MLA_7756710937241
// MI455X (gfx1250) — matrix operand order repaired and checked by reading, not compiled, not run on hardware
//
#include <hip/hip_runtime.h>
#include <hip/hip_bf16.h>
#include <hip/hip_version.h>
#include <cstdint>
#include <cstddef>

// ---------------- constants ----------------
#define TOKENS 2048            // B*S
#define SEQ    1024
#define NHEADS 32
#define DMODEL 4096
#define QLR    1536
#define KVLR   512
#define DNOPE  128
#define DROPE  64
#define DQKH   192             // NOPE+ROPE
#define DV     128
#define ATT_SCALE 0.07216878364870323f   // 192^-0.5
#define MASK_NEG  (-1.0e9f)

typedef __attribute__((ext_vector_type(16))) __bf16 bf16x16;
typedef __attribute__((ext_vector_type(8)))  __bf16 bf16x8;
typedef __attribute__((ext_vector_type(8)))  float  f32x8;
typedef unsigned int v4u __attribute__((ext_vector_type(4)));
typedef int          v8i __attribute__((ext_vector_type(8)));
typedef int          v4i __attribute__((ext_vector_type(4)));

// ---------------- TDM availability ----------------
#if defined(__HIP_DEVICE_COMPILE__) && defined(__has_builtin)
#if __has_builtin(__builtin_amdgcn_tensor_load_to_lds) && \
    __has_builtin(__builtin_amdgcn_s_wait_tensorcnt)
#define MLA_HAVE_TDM 1
#endif
#endif
#ifndef MLA_HAVE_TDM
#define MLA_HAVE_TDM 0
#endif

#if MLA_HAVE_TDM
// This toolchain's builtin takes 6 args: (g0, g1, g2, g3, g4, cpol)
#define MLA_TDM_CALL(g0, g1, g2, g3) \
  __builtin_amdgcn_tensor_load_to_lds(g0, g1, g2, g3, (v8i){0, 0, 0, 0, 0, 0, 0, 0}, 0)

// 2D tile DMA: global row-major [rows][stride] bf16 -> LDS rows padded to 40 elems.
// pad_interval code 3 = every 16 DWORDs (one 32-elem bf16 row), pad_amount code 3 = 4 DWORDs.
static __device__ __forceinline__ void tdm_load_2d(unsigned lds_addr, const void* gaddr,
                                                   int tile_rows, int stride_elems) {
  unsigned long long ga = (unsigned long long)(uintptr_t)gaddr;
  v4u g0;
  g0.x = 1u;                                              // count=1, user mode
  g0.y = lds_addr;                                        // LDS byte address
  g0.z = (unsigned)(ga & 0xffffffffull);                  // global addr [31:0]
  g0.w = (unsigned)((ga >> 32) & 0x01ffffffull) | 0x80000000u;  // [56:32] | type=2
  v8i g1;
  g1[0] = (1 << 16) | (1 << 20) | (3 << 22) | (3 << 25);  // 2B elems, pad 16DW/+4DW
  unsigned td0 = 0x7fffffffu, td1 = 0x7fffffffu;          // generous tensor dims (no OOB)
  g1[1] = (int)((td0 & 0xffffu) << 16);                   // tensor_dim0[15:0] @ bits 63:48
  g1[2] = (int)((td0 >> 16) | ((td1 & 0xffffu) << 16));   // dim0[31:16] | dim1[15:0]
  g1[3] = (int)((td1 >> 16) | (32u << 16));               // dim1[31:16] | tile_dim0=32
  g1[4] = tile_rows;                                      // tile_dim1 (tile_dim2=0)
  g1[5] = stride_elems;                                   // tensor_dim0_stride lo32
  g1[6] = 0;
  g1[7] = 0;
  v4i z4 = {0, 0, 0, 0};
  MLA_TDM_CALL(g0, g1, z4, z4);
}
#endif  // MLA_HAVE_TDM

// ---------------- WMMA helpers ----------------
static __device__ __forceinline__ f32x8 zero8() {
  f32x8 z;
#pragma unroll
  for (int i = 0; i < 8; ++i) z[i] = 0.0f;
  return z;
}

static __device__ __forceinline__ bf16x16 make_frag(const __bf16* p0, const __bf16* p1) {
  bf16x8 a = *(const bf16x8*)p0;
  bf16x8 b = *(const bf16x8*)p1;
  bf16x16 r;
#pragma unroll
  for (int i = 0; i < 8; ++i) { r[i] = a[i]; r[8 + i] = b[i]; }
  return r;
}

// A fragment (16 rows x 32 K) from row-major [rows][ld], base at (row0, 0).
// ISA: lanes 0-15 hold M=lane, K={0..7,16..23}; lanes 16-31 K={8..15,24..31}.
static __device__ __forceinline__ bf16x16 frag_a(const __bf16* base, int ld, int k0, int lane) {
  const __bf16* p = base + (size_t)(lane & 15) * ld + k0 + ((lane & 16) ? 8 : 0);
  return make_frag(p, p + 16);
}

// B fragment (32 K x 16 N) from N-major [n][ld over K], base at (n0, 0).
static __device__ __forceinline__ bf16x16 frag_b(const __bf16* base, int ld, int k0, int lane) {
  const __bf16* p = base + (size_t)(lane & 15) * ld + k0 + ((lane & 16) ? 8 : 0);
  return make_frag(p, p + 16);
}

static __device__ __forceinline__ f32x8 wmma_bf16(bf16x16 a, bf16x16 b, f32x8 c) {
  return __builtin_amdgcn_wmma_f32_16x16x32_bf16(false, a, false, b, (short)0, c, false, false);
}

// ---------------- fp32 -> bf16 conversion ----------------
__global__ __launch_bounds__(256) void mla_cvt_bf16(const float* __restrict__ in,
                                                    __bf16* __restrict__ out, size_t n) {
  size_t i = (size_t)blockIdx.x * blockDim.x + threadIdx.x;
  size_t stride = (size_t)gridDim.x * blockDim.x;
  for (; i < n; i += stride) out[i] = (__bf16)in[i];
}

// ---------------- fp32 [R][C] -> bf16 [C][R] transpose (32x32 LDS tiles) -------------
__global__ __launch_bounds__(256) void mla_cvt_t(const float* __restrict__ in, long long sin,
                                                 __bf16* __restrict__ out, long long sout,
                                                 int R, int C) {
  __shared__ float t[32][33];
  in  += (size_t)blockIdx.z * sin;
  out += (size_t)blockIdx.z * sout;
  int c0 = blockIdx.x * 32, r0 = blockIdx.y * 32;
  int tx = threadIdx.x & 31, ty = threadIdx.x >> 5;
#pragma unroll
  for (int i = 0; i < 32; i += 8)
    t[ty + i][tx] = in[(size_t)(r0 + ty + i) * C + c0 + tx];
  __syncthreads();
#pragma unroll
  for (int i = 0; i < 32; i += 8)
    out[(size_t)(c0 + ty + i) * R + r0 + tx] = (__bf16)t[tx][ty + i];
}

// ---------------- RMSNorm (fp32 in, bf16 out) ----------------
__global__ __launch_bounds__(256) void mla_rmsnorm(const float* __restrict__ in,
                                                   const float* __restrict__ w,
                                                   __bf16* __restrict__ out,
                                                   int C, int in_stride) {
  __shared__ float red[8];
  int row = blockIdx.x;
  const float* x = in + (size_t)row * in_stride;
  float ss = 0.0f;
  for (int c = threadIdx.x; c < C; c += 256) { float v = x[c]; ss += v * v; }
#pragma unroll
  for (int off = 16; off > 0; off >>= 1) ss += __shfl_xor(ss, off, 32);
  if ((threadIdx.x & 31) == 0) red[threadIdx.x >> 5] = ss;
  __syncthreads();
  if (threadIdx.x == 0) {
    float t = 0.0f;
#pragma unroll
    for (int i = 0; i < 8; ++i) t += red[i];
    red[0] = rsqrtf(t / (float)C + 1e-6f);
  }
  __syncthreads();
  float rn = red[0];
  for (int c = threadIdx.x; c < C; c += 256)
    out[(size_t)row * C + c] = (__bf16)(x[c] * rn * w[c]);
}

// ---------------- RoPE + split q (fp32 q -> head-major bf16) ----------------
__global__ __launch_bounds__(128) void mla_rope_q(const float* __restrict__ q,     // [2048][H*192]
                                                  const float* __restrict__ freqs, // [1024][32][2]
                                                  __bf16* __restrict__ qnope,      // [H][2048][128]
                                                  __bf16* __restrict__ qpe) {      // [H][2048][64]
  int tok = blockIdx.x;
  int h   = blockIdx.y;
  int s   = tok & (SEQ - 1);
  const float* qh = q + (size_t)tok * (NHEADS * DQKH) + h * DQKH;
  int t = threadIdx.x;
  qnope[((size_t)h * TOKENS + tok) * DNOPE + t] = (__bf16)qh[t];
  if (t < 32) {
    float xe = qh[DNOPE + 2 * t];
    float xo = qh[DNOPE + 2 * t + 1];
    float cs = freqs[((size_t)s * 32 + t) * 2 + 0];
    float sn = freqs[((size_t)s * 32 + t) * 2 + 1];
    __bf16* o = qpe + ((size_t)h * TOKENS + tok) * DROPE;
    o[2 * t]     = (__bf16)(xe * cs - xo * sn);
    o[2 * t + 1] = (__bf16)(xe * sn + xo * cs);
  }
}

// ---------------- RoPE k_pe (slice of kv fp32) ----------------
__global__ __launch_bounds__(32) void mla_rope_k(const float* __restrict__ kv,    // [2048][576]
                                                 const float* __restrict__ freqs,
                                                 __bf16* __restrict__ kpe) {      // [2048][64]
  int tok = blockIdx.x;
  int s   = tok & (SEQ - 1);
  int t = threadIdx.x;
  float xe = kv[(size_t)tok * (KVLR + DROPE) + KVLR + 2 * t];
  float xo = kv[(size_t)tok * (KVLR + DROPE) + KVLR + 2 * t + 1];
  float cs = freqs[((size_t)s * 32 + t) * 2 + 0];
  float sn = freqs[((size_t)s * 32 + t) * 2 + 1];
  kpe[(size_t)tok * DROPE + 2 * t]     = (__bf16)(xe * cs - xo * sn);
  kpe[(size_t)tok * DROPE + 2 * t + 1] = (__bf16)(xe * sn + xo * cs);
}

// ---------------- bf16 WMMA GEMM: C[M][N] = A[M][K] * B^T, B stored [N][K] -----------
// Block tile 128x64, BK=32, 8 waves: wave w owns rows w*16, all 64 cols (4 wmma/step).
// TDM path: double-buffered tensor_load_to_lds DMAs overlapped with WMMA.
__global__ __launch_bounds__(256) void mla_gemm_bf16(
    const __bf16* __restrict__ A, long long sA, int lda,
    const __bf16* __restrict__ B, long long sB, int ldb,
    void* __restrict__ C, long long sC, int ldc, int c_is_bf16,
    int M, int N, int K) {
  __shared__ __bf16 lA[2][128][40];
  __shared__ __bf16 lB[2][64][40];

  int bat = blockIdx.z;
  A += (size_t)bat * sA;
  B += (size_t)bat * sB;
  int m0 = blockIdx.y * 128, n0 = blockIdx.x * 64;
  int lane = threadIdx.x & 31;
  int wid  = threadIdx.x >> 5;

  f32x8 acc[4];
#pragma unroll
  for (int nt = 0; nt < 4; ++nt) acc[nt] = zero8();

  int steps = K >> 5;

#if MLA_HAVE_TDM
  const __bf16* Abase = A + (size_t)m0 * lda;
  const __bf16* Bbase = B + (size_t)n0 * ldb;
  if (wid == 0) {
    tdm_load_2d((unsigned)(uintptr_t)&lA[0][0][0], Abase, 128, lda);
    tdm_load_2d((unsigned)(uintptr_t)&lB[0][0][0], Bbase, 64, ldb);
  }
  for (int i = 0; i < steps; ++i) {
    if (wid == 0) {
      if (i + 1 < steps) {
        int kb = (i + 1) * 32;
        tdm_load_2d((unsigned)(uintptr_t)&lA[(i + 1) & 1][0][0], Abase + kb, 128, lda);
        tdm_load_2d((unsigned)(uintptr_t)&lB[(i + 1) & 1][0][0], Bbase + kb, 64, ldb);
        __builtin_amdgcn_s_wait_tensorcnt(2);   // current pair landed
      } else {
        __builtin_amdgcn_s_wait_tensorcnt(0);
      }
    }
    __syncthreads();
    const __bf16* bufA = &lA[i & 1][0][0];
    const __bf16* bufB = &lB[i & 1][0][0];
    bf16x16 af = frag_a(bufA + wid * 16 * 40, 40, 0, lane);
#pragma unroll
    for (int nt = 0; nt < 4; ++nt) {
      bf16x16 bf_ = frag_b(bufB + nt * 16 * 40, 40, 0, lane);
      acc[nt] = wmma_bf16(af, bf_, acc[nt]);
    }
    __syncthreads();
  }
#else
  for (int i = 0; i < steps; ++i) {
    int k0 = i * 32;
    for (int t = threadIdx.x; t < (128 * 32) / 8; t += 256) {
      int idx = t * 8, r = idx >> 5, c = idx & 31;
      *(bf16x8*)&lA[0][r][c] = *(const bf16x8*)(A + (size_t)(m0 + r) * lda + k0 + c);
    }
    {
      int idx = threadIdx.x * 8, r = idx >> 5, c = idx & 31;
      *(bf16x8*)&lB[0][r][c] = *(const bf16x8*)(B + (size_t)(n0 + r) * ldb + k0 + c);
    }
    __syncthreads();
    bf16x16 af = frag_a(&lA[0][wid * 16][0], 40, 0, lane);
#pragma unroll
    for (int nt = 0; nt < 4; ++nt) {
      bf16x16 bf_ = frag_b(&lB[0][nt * 16][0], 40, 0, lane);
      acc[nt] = wmma_bf16(af, bf_, acc[nt]);
    }
    __syncthreads();
  }
#endif

  int mrow = m0 + wid * 16 + ((lane & 16) ? 8 : 0);
  int ncb  = n0 + (lane & 15);
#pragma unroll
  for (int nt = 0; nt < 4; ++nt) {
#pragma unroll
    for (int v = 0; v < 8; ++v) {
      size_t idx = (size_t)bat * sC + (size_t)(mrow + v) * ldc + ncb + nt * 16;
      if (c_is_bf16) ((__bf16*)C)[idx] = (__bf16)acc[nt][v];
      else           ((float*)C)[idx]  = acc[nt][v];
    }
  }
}

// ---------------- flash attention (per (b,h), 32 query rows / block) ----------------
__global__ __launch_bounds__(256) void mla_attn(
    const __bf16* __restrict__ qabs,  // [H][2048][512]
    const __bf16* __restrict__ qpe,   // [H][2048][64]
    const __bf16* __restrict__ kvn,   // [2048][512]
    const __bf16* __restrict__ kpe,   // [2048][64]
    __bf16* __restrict__ o_attn) {    // [H][2048][512]
  __shared__ float  sS[32][72];
  __shared__ __bf16 sP[32][80];
  __shared__ __bf16 sVT[512][40];
  __shared__ float  sM[32], sL[32], sAlpha[32];

  int b  = blockIdx.z;
  int h  = blockIdx.y;
  int m0 = blockIdx.x * 32;
  int tokb = b * SEQ;
  int lane = threadIdx.x & 31;
  int wid  = threadIdx.x >> 5;

  const __bf16* Qa = qabs + ((size_t)h * TOKENS + tokb + m0) * KVLR;
  const __bf16* Qp = qpe  + ((size_t)h * TOKENS + tokb + m0) * DROPE;
  const __bf16* Kn = kvn + (size_t)tokb * KVLR;
  const __bf16* Kp = kpe + (size_t)tokb * DROPE;

  if (threadIdx.x < 32) { sM[threadIdx.x] = -3.0e38f; sL[threadIdx.x] = 0.0f; }

  f32x8 acc[2][4];
#pragma unroll
  for (int mt = 0; mt < 2; ++mt)
#pragma unroll
    for (int nt = 0; nt < 4; ++nt) acc[mt][nt] = zero8();

  int sm0 = (wid & 1) * 16;
  int sn0 = (wid >> 1) * 16;
  int jt_max = (m0 + 31) >> 6;

  for (int jt = 0; jt <= jt_max; ++jt) {
    int k0key = jt * 64;

    f32x8 sacc = zero8();
    for (int kk = 0; kk < KVLR; kk += 32) {
      bf16x16 af  = frag_a(Qa + (size_t)sm0 * KVLR, KVLR, kk, lane);
      bf16x16 bf_ = frag_b(Kn + (size_t)(k0key + sn0) * KVLR, KVLR, kk, lane);
      sacc = wmma_bf16(af, bf_, sacc);
    }
#pragma unroll
    for (int kk = 0; kk < DROPE; kk += 32) {
      bf16x16 af  = frag_a(Qp + (size_t)sm0 * DROPE, DROPE, kk, lane);
      bf16x16 bf_ = frag_b(Kp + (size_t)(k0key + sn0) * DROPE, DROPE, kk, lane);
      sacc = wmma_bf16(af, bf_, sacc);
    }
    {
      int mm = sm0 + ((lane & 16) ? 8 : 0);
      int nn = sn0 + (lane & 15);
#pragma unroll
      for (int v = 0; v < 8; ++v) {
        float val = sacc[v] * ATT_SCALE;
        if (k0key + nn > m0 + mm + v) val += MASK_NEG;
        sS[mm + v][nn] = val;
      }
    }
    __syncthreads();

    if (wid == 0) {
      int r = lane;
      float mold = sM[r];
      float mnew = mold;
      for (int c = 0; c < 64; ++c) mnew = fmaxf(mnew, sS[r][c]);
      float alpha = __expf(mold - mnew);
      float lsum = 0.0f;
      for (int c = 0; c < 64; ++c) {
        float pv = __expf(sS[r][c] - mnew);
        lsum += pv;
        sP[r][c] = (__bf16)pv;
      }
      sM[r] = mnew;
      sL[r] = sL[r] * alpha + lsum;
      sAlpha[r] = alpha;
    }
    __syncthreads();

#pragma unroll
    for (int mt = 0; mt < 2; ++mt) {
      int mmb = mt * 16 + ((lane & 16) ? 8 : 0);
#pragma unroll
      for (int v = 0; v < 8; ++v) {
        float a = sAlpha[mmb + v];
#pragma unroll
        for (int nt = 0; nt < 4; ++nt) acc[mt][nt][v] *= a;
      }
    }

#pragma unroll
    for (int kk = 0; kk < 64; kk += 32) {
      for (int idx = threadIdx.x; idx < 32 * KVLR; idx += 256) {
        int kp = idx >> 9;
        int n  = idx & 511;
        sVT[n][kp] = Kn[(size_t)(k0key + kk + kp) * KVLR + n];
      }
      __syncthreads();
#pragma unroll
      for (int mt = 0; mt < 2; ++mt) {
        bf16x16 pf = frag_a(&sP[mt * 16][0], 80, kk, lane);
#pragma unroll
        for (int nt = 0; nt < 4; ++nt) {
          bf16x16 vf = frag_b(&sVT[wid * 64 + nt * 16][0], 40, 0, lane);
          acc[mt][nt] = wmma_bf16(pf, vf, acc[mt][nt]);
        }
      }
      __syncthreads();
    }
  }

#pragma unroll
  for (int mt = 0; mt < 2; ++mt) {
    int mmb = mt * 16 + ((lane & 16) ? 8 : 0);
#pragma unroll
    for (int nt = 0; nt < 4; ++nt) {
      int nn = wid * 64 + nt * 16 + (lane & 15);
#pragma unroll
      for (int v = 0; v < 8; ++v) {
        float o = acc[mt][nt][v] / sL[mmb + v];
        o_attn[((size_t)h * TOKENS + tokb + m0 + mmb + v) * KVLR + nn] = (__bf16)o;
      }
    }
  }
}

// ---------------- launch ----------------
extern "C" void kernel_launch(void* const* d_in, const int* in_sizes, int n_in,
                              void* d_out, int out_size, void* d_ws, size_t ws_size,
                              hipStream_t stream) {
  const float* x     = (const float*)d_in[0];
  const float* freqs = (const float*)d_in[1];
  const float* wq_a  = (const float*)d_in[3];
  const float* q_nw  = (const float*)d_in[4];
  const float* wq_b  = (const float*)d_in[5];
  const float* wkv_a = (const float*)d_in[6];
  const float* kv_nw = (const float*)d_in[7];
  const float* wkv_b = (const float*)d_in[8];
  const float* wo    = (const float*)d_in[9];
  float* out = (float*)d_out;

  char* p = (char*)d_ws;
  auto abf  = [&](size_t e) { __bf16* r = (__bf16*)p; p += e * sizeof(__bf16); return r; };
  auto af32 = [&](size_t e) { float* r = (float*)p; p += e * sizeof(float); return r; };

  __bf16* xbf     = abf((size_t)TOKENS * DMODEL);
  __bf16* wqaT    = abf((size_t)QLR * DMODEL);                    // [N][K]
  __bf16* wqbT    = abf((size_t)(NHEADS * DQKH) * QLR);           // [N][K]
  __bf16* wkvaT   = abf((size_t)(KVLR + DROPE) * DMODEL);         // [N][K]
  __bf16* wkvbT1  = abf((size_t)NHEADS * KVLR * DNOPE);           // [h][c=512][d=128]
  __bf16* wkvb_bf = abf((size_t)NHEADS * (DNOPE + DV) * KVLR);    // plain copy
  __bf16* woT     = abf((size_t)DMODEL * (NHEADS * DV));          // [N][K]
  __bf16* qn_bf   = abf((size_t)TOKENS * QLR);
  __bf16* kvn_bf  = abf((size_t)TOKENS * KVLR);
  __bf16* kpe_bf  = abf((size_t)TOKENS * DROPE);
  __bf16* qnope   = abf((size_t)NHEADS * TOKENS * DNOPE);
  __bf16* qpe_bf  = abf((size_t)NHEADS * TOKENS * DROPE);
  __bf16* qabs    = abf((size_t)NHEADS * TOKENS * KVLR);
  __bf16* attnb   = abf((size_t)NHEADS * TOKENS * KVLR);
  __bf16* outtmp  = abf((size_t)TOKENS * DMODEL);
  float* qa_f32 = af32((size_t)TOKENS * QLR);
  float* q_f32  = af32((size_t)TOKENS * (NHEADS * DQKH));
  float* kv_f32 = af32((size_t)TOKENS * (KVLR + DROPE));

  auto cvt = [&](const float* src, __bf16* dst, size_t n) {
    int g = (int)((n + 2047) / 2048); if (g > 65535) g = 65535; if (g < 1) g = 1;
    mla_cvt_bf16<<<dim3(g), dim3(256), 0, stream>>>(src, dst, n);
  };
  auto cvtT = [&](const float* src, long long sin, __bf16* dst, long long sout,
                  int R, int C, int batch) {
    dim3 g(C / 32, R / 32, batch);
    mla_cvt_t<<<g, dim3(256), 0, stream>>>(src, sin, dst, sout, R, C);
  };
  auto gemm = [&](const __bf16* A, long long sa, int lda,
                  const __bf16* B, long long sb, int ldb,
                  void* C, long long sc, int ldc, int cbf,
                  int M, int N, int K, int batch) {
    dim3 g(N / 64, M / 128, batch);
    mla_gemm_bf16<<<g, dim3(256), 0, stream>>>(A, sa, lda, B, sb, ldb, C, sc, ldc, cbf, M, N, K);
  };

  // 0) conversions / weight transposes (one-time, bandwidth-bound)
  cvt(x, xbf, (size_t)TOKENS * DMODEL);
  cvt(wkv_b, wkvb_bf, (size_t)NHEADS * (DNOPE + DV) * KVLR);
  cvtT(wq_a, 0, wqaT, 0, DMODEL, QLR, 1);
  cvtT(wkv_a, 0, wkvaT, 0, DMODEL, KVLR + DROPE, 1);
  cvtT(wq_b, 0, wqbT, 0, QLR, NHEADS * DQKH, 1);
  cvtT(wo, 0, woT, 0, NHEADS * DV, DMODEL, 1);
  cvtT(wkv_b, (long long)(DNOPE + DV) * KVLR, wkvbT1, (long long)KVLR * DNOPE,
       DNOPE, KVLR, NHEADS);

  // 1) q_a = x @ wq_a ; kv = x @ wkv_a (fp32 out for rmsnorm)
  gemm(xbf, 0, DMODEL, wqaT, 0, DMODEL, qa_f32, 0, QLR, 0, TOKENS, QLR, DMODEL, 1);
  gemm(xbf, 0, DMODEL, wkvaT, 0, DMODEL, kv_f32, 0, KVLR + DROPE, 0,
       TOKENS, KVLR + DROPE, DMODEL, 1);

  // 2) rmsnorms (-> bf16)
  mla_rmsnorm<<<dim3(TOKENS), dim3(256), 0, stream>>>(qa_f32, q_nw, qn_bf, QLR, QLR);
  mla_rmsnorm<<<dim3(TOKENS), dim3(256), 0, stream>>>(kv_f32, kv_nw, kvn_bf, KVLR, KVLR + DROPE);

  // 3) q = qn @ wq_b (fp32), rope + head-major split
  gemm(qn_bf, 0, QLR, wqbT, 0, QLR, q_f32, 0, NHEADS * DQKH, 0,
       TOKENS, NHEADS * DQKH, QLR, 1);
  mla_rope_q<<<dim3(TOKENS, NHEADS), dim3(128), 0, stream>>>(q_f32, freqs, qnope, qpe_bf);
  mla_rope_k<<<dim3(TOKENS), dim3(32), 0, stream>>>(kv_f32, freqs, kpe_bf);

  // 4) q_abs[h] = qnope[h] @ wkv_b[h,:128,:]  (B pre-transposed per head)
  gemm(qnope, (long long)TOKENS * DNOPE, DNOPE,
       wkvbT1, (long long)KVLR * DNOPE, DNOPE,
       qabs, (long long)TOKENS * KVLR, KVLR, 1,
       TOKENS, KVLR, DNOPE, NHEADS);

  // 5) flash attention
  mla_attn<<<dim3(SEQ / 32, NHEADS, 2), dim3(256), 0, stream>>>(qabs, qpe_bf, kvn_bf,
                                                                kpe_bf, attnb);

  // 6) per-head V-projection: B = wkv_b[h,128:,:] already [N=d][K=c]
  gemm(attnb, (long long)TOKENS * KVLR, KVLR,
       wkvb_bf + (size_t)DNOPE * KVLR, (long long)(DNOPE + DV) * KVLR, KVLR,
       outtmp, (long long)DV, NHEADS * DV, 1,
       TOKENS, DV, KVLR, NHEADS);

  // 7) final projection (fp32 out)
  gemm(outtmp, 0, NHEADS * DV, woT, 0, NHEADS * DV, out, 0, DMODEL, 0,
       TOKENS, DMODEL, NHEADS * DV, 1);
}
